// SSMBlock_43207370997794
// MI455X (gfx1250) — matrix operand order repaired and checked by reading, not compiled, not run on hardware
//
#include <hip/hip_runtime.h>

#define LN_EPS 0.001f

typedef __bf16 v16bf __attribute__((ext_vector_type(16)));
typedef float  v8f   __attribute__((ext_vector_type(8)));

union Frag16 { v16bf v; uint4 q[2]; };
union Acc8   { v8f v; float f[8]; };

static __device__ __forceinline__ unsigned short f32_to_bf16_rne(float f) {
    unsigned int u = __float_as_uint(f);
    unsigned int lsb = (u >> 16) & 1u;
    u += 0x7fffu + lsb;                 // round-to-nearest-even
    return (unsigned short)(u >> 16);
}

// ---------------------------------------------------------------------------
// Kernel 1: LayerNorm over C=96, output bf16 row-major [M][96].
// One wave (32 lanes) per row; 3 elements per lane; wave32 shuffle reduce.
// ---------------------------------------------------------------------------
__global__ __launch_bounds__(256) void ln_to_bf16(
    const float* __restrict__ x,
    const float* __restrict__ gamma,
    const float* __restrict__ beta,
    unsigned short* __restrict__ xn)
{
    const int C = 96;
    int row  = blockIdx.x * 8 + (threadIdx.x >> 5);
    int lane = threadIdx.x & 31;
    int c0   = lane * 3;

    const float* xr = x + (long long)row * C + c0;
    float v0 = xr[0], v1 = xr[1], v2 = xr[2];

    float s = v0 + v1 + v2;
    #pragma unroll
    for (int m = 1; m < 32; m <<= 1) s += __shfl_xor(s, m);
    float mu = s * (1.0f / 96.0f);

    float d0 = v0 - mu, d1 = v1 - mu, d2 = v2 - mu;
    float q = d0 * d0 + d1 * d1 + d2 * d2;
    #pragma unroll
    for (int m = 1; m < 32; m <<= 1) q += __shfl_xor(q, m);
    float rs = rsqrtf(q * (1.0f / 96.0f) + LN_EPS);

    unsigned short* o = xn + (long long)row * C + c0;
    o[0] = f32_to_bf16_rne(d0 * rs * gamma[c0 + 0] + beta[c0 + 0]);
    o[1] = f32_to_bf16_rne(d1 * rs * gamma[c0 + 1] + beta[c0 + 1]);
    o[2] = f32_to_bf16_rne(d2 * rs * gamma[c0 + 2] + beta[c0 + 2]);
}

// ---------------------------------------------------------------------------
// Kernel 2: transpose W_in [96][4896] -> winT bf16 [4896][96]
//           transpose W_out [96][96]  -> woutT bf16 [96][96]
// grid*block == 4896*96 + 96*96 exactly.
// ---------------------------------------------------------------------------
__global__ __launch_bounds__(256) void convert_weights(
    const float* __restrict__ W_in,
    const float* __restrict__ W_out,
    unsigned short* __restrict__ winT,
    unsigned short* __restrict__ woutT)
{
    const int C = 96, TOT = 4896;
    int idx = blockIdx.x * 256 + threadIdx.x;
    int nIn = TOT * C;                              // 470016
    if (idx < nIn) {
        int n = idx / C, k = idx - n * C;
        winT[idx] = f32_to_bf16_rne(W_in[(long long)k * TOT + n]);
    } else {
        int j = idx - nIn;                          // < 9216
        int n = j / C, k = j - n * C;
        woutT[j] = f32_to_bf16_rne(W_out[k * C + n]);
    }
}

// ---------------------------------------------------------------------------
// Kernel 3: bf16 WMMA GEMM  out[M][N] = A[M][96] * Bt[N][96]^T + bias[N]
// Block = 128 threads (4 waves), tile = 128(M) x 32(N), K = 96 (3 x k32).
// Waves arranged 2(M) x 2(N); each wave computes 64x16 via 4 accumulators.
// Fragment layouts (contiguous k order, same for A and B):
//   A (16x32 bf16): lanes m=lane&15, khalf=lane>>4;
//     elems 0..15 <- K = kk*32 + khalf*16 + 0..15    (two b128 LDS loads)
//   B (32x16 bf16): lanes n=lane&15, khalf=lane>>4;
//     elems 0..15 <- K = kk*32 + khalf*16 + 0..15    (two b128 LDS loads)
//   C/D (16x16 f32): VGPR r -> M = r + 8*khalf, N = lane&15.
// ---------------------------------------------------------------------------
__global__ __launch_bounds__(128) void gemm_bf16_wmma(
    const unsigned short* __restrict__ A,    // [M][96] bf16
    const unsigned short* __restrict__ Bt,   // [N][96] bf16 (transposed weights)
    const float* __restrict__ bias,          // [N]
    float* __restrict__ out,                 // [M][N] f32
    int N)
{
    const int K = 96;
    __shared__ __align__(16) unsigned short lA[128 * 96];   // 24 KB
    __shared__ __align__(16) unsigned short lB[32 * 96];    //  6 KB

    int tid = threadIdx.x;
    long long m0 = (long long)blockIdx.y * 128;
    int n0 = blockIdx.x * 32;

    // Stage A tile: first 96 B (6 x b128) of one row per thread.
    {
        const uint4* src = (const uint4*)(A + (m0 + tid) * K);
        uint4* dst = (uint4*)(lA + tid * K);
        #pragma unroll
        for (int i = 0; i < 6; ++i) dst[i] = src[i];
    }
    // Stage Bt tile: 4 threads per row, 48 B (3 x b128) each.
    {
        int r = tid >> 2, ch = tid & 3;
        const uint4* src = (const uint4*)(Bt + (n0 + r) * K + ch * 24);
        uint4* dst = (uint4*)(lB + r * K + ch * 24);
        #pragma unroll
        for (int i = 0; i < 3; ++i) dst[i] = src[i];
    }
    __syncthreads();

    int wave  = tid >> 5;
    int lane  = tid & 31;
    int waveM = wave >> 1;        // 0..1 -> 64 rows each
    int waveN = wave & 1;         // 0..1 -> 16 cols each
    int l16   = lane & 15;
    int khalf = lane >> 4;

    v8f acc[4] = {};

    #pragma unroll
    for (int kk = 0; kk < 3; ++kk) {
        Frag16 bfrag;
        const unsigned short* bp = lB + (waveN * 16 + l16) * K + kk * 32 + khalf * 16;
        bfrag.q[0] = *(const uint4*)bp;
        bfrag.q[1] = *(const uint4*)(bp + 8);
        #pragma unroll
        for (int mt = 0; mt < 4; ++mt) {
            Frag16 afrag;
            const unsigned short* ap =
                lA + (waveM * 64 + mt * 16 + l16) * K + kk * 32 + khalf * 16;
            afrag.q[0] = *(const uint4*)ap;
            afrag.q[1] = *(const uint4*)(ap + 8);
            acc[mt] = __builtin_amdgcn_wmma_f32_16x16x32_bf16(
                false, afrag.v, false, bfrag.v, (short)0, acc[mt], false, false);
        }
    }

    int ng = n0 + waveN * 16 + l16;
    float bv = bias[ng];
    #pragma unroll
    for (int mt = 0; mt < 4; ++mt) {
        Acc8 u; u.v = acc[mt];
        long long mb = m0 + waveM * 64 + mt * 16 + khalf * 8;
        #pragma unroll
        for (int r = 0; r < 8; ++r)
            out[(mb + r) * (long long)N + ng] = u.f[r] + bv;
    }
}

// ---------------------------------------------------------------------------
// Kernel 4: selective-scan. 384 independent (b,c) channels, S=16 states.
// Each wave handles 2 channels: lanes 0-15 and 16-31 hold the 16 states.
// fp32 recurrence; 16-lane xor-shuffle reduction for sum_s(C*h); sigmoid gate
// fused; y written as bf16 for the output GEMM. Prefetch 8 rows ahead.
// ---------------------------------------------------------------------------
__global__ __launch_bounds__(256) void ssm_scan(
    const float* __restrict__ proj,          // [B*L][4896] f32 (bias included)
    unsigned short* __restrict__ ybf)        // [B*L][96] bf16
{
    const int C = 96, S = 16, TOT = 4896, L = 4096;
    int wave = threadIdx.x >> 5;
    int lane = threadIdx.x & 31;
    int chan = (blockIdx.x * 8 + wave) * 2 + (lane >> 4);  // 0..383
    int s    = lane & 15;
    int b    = chan / C;
    int c    = chan - b * C;

    int offV = c;
    int offG = C + c;
    int offD = 2 * C + c;
    int offA = 3 * C + c * S + s;
    int offB = 3 * C + C * S + c * S + s;
    int offC = 3 * C + 2 * C * S + c * S + s;

    long long row = (long long)b * L * TOT;
    float h = 0.0f;

    for (int l = 0; l < L; ++l, row += TOT) {
        if (l + 8 < L)   // speculative prefetch of a future timestep row
            __builtin_prefetch(proj + row + 8LL * TOT + offA, 0, 0);

        float a  = proj[row + offA];
        float bm = proj[row + offB];
        float cm = proj[row + offC];
        float dl = proj[row + offD];
        float u  = proj[row + offV];
        float gr = proj[row + offG];

        h = dl * (a * h) + dl * (bm * u);

        float p = cm * h;
        p += __shfl_xor(p, 1);
        p += __shfl_xor(p, 2);
        p += __shfl_xor(p, 4);
        p += __shfl_xor(p, 8);

        if (s == 0) {
            float g = 1.0f / (1.0f + __expf(-gr));
            ybf[((long long)b * L + l) * C + c] = f32_to_bf16_rne(g * p);
        }
    }
}

// ---------------------------------------------------------------------------
// Launch
// ---------------------------------------------------------------------------
extern "C" void kernel_launch(void* const* d_in, const int* in_sizes, int n_in,
                              void* d_out, int out_size, void* d_ws, size_t ws_size,
                              hipStream_t stream) {
    (void)in_sizes; (void)n_in; (void)out_size; (void)ws_size;

    const float* x     = (const float*)d_in[0];
    const float* gamma = (const float*)d_in[1];
    const float* beta  = (const float*)d_in[2];
    const float* W_in  = (const float*)d_in[3];
    const float* b_in  = (const float*)d_in[4];
    const float* W_out = (const float*)d_in[5];
    const float* b_out = (const float*)d_in[6];
    float* out = (float*)d_out;

    const int M   = 4 * 64 * 64;   // 16384 tokens
    const int C   = 96;
    const int TOT = 4896;          // 3C + 3CS

    // Carve workspace (256B aligned regions).
    char* ws = (char*)d_ws;
    size_t off = 0;
    auto carve = [&](size_t bytes) -> void* {
        void* p = ws + off;
        off += (bytes + 255) & ~(size_t)255;
        return p;
    };
    unsigned short* xn    = (unsigned short*)carve((size_t)M * C * 2);      // 3.1 MB
    unsigned short* winT  = (unsigned short*)carve((size_t)TOT * C * 2);    // 0.9 MB
    unsigned short* woutT = (unsigned short*)carve((size_t)C * C * 2);      // 18 KB
    unsigned short* ybf   = (unsigned short*)carve((size_t)M * C * 2);      // 3.1 MB
    float*          proj  = (float*)carve((size_t)M * TOT * 4);             // 321 MB

    // 1) LayerNorm -> bf16 activations
    ln_to_bf16<<<M / 8, 256, 0, stream>>>(x, gamma, beta, xn);

    // 2) Weights -> transposed bf16
    convert_weights<<<(TOT * C + C * C) / 256, 256, 0, stream>>>(W_in, W_out, winT, woutT);

    // 3) proj = xn @ W_in + b_in   (16384 x 4896, K=96), WMMA bf16
    gemm_bf16_wmma<<<dim3(TOT / 32, M / 128), 128, 0, stream>>>(xn, winT, b_in, proj, TOT);

    // 4) selective scan -> y (bf16)
    ssm_scan<<<24, 256, 0, stream>>>(proj, ybf);

    // 5) out = y @ W_out + b_out   (16384 x 96, K=96), WMMA bf16
    gemm_bf16_wmma<<<dim3(C / 32, M / 128), 128, 0, stream>>>(ybf, woutT, b_out, out, C);
}
